// UnsupervisedAnomalyDetectionModel_17617955848308
// MI455X (gfx1250) — matrix operand order repaired and checked by reading, not compiled, not run on hardware
//
#include <hip/hip_runtime.h>
#include <hip/hip_bf16.h>

typedef __attribute__((ext_vector_type(16))) _Float16 v16h;
typedef __attribute__((ext_vector_type(8)))  float    v8f;

#define BATCH   65536
#define KG      60        // groups
#define CW      16        // per-group width
#define HW      64        // hidden
#define WAVES   4         // waves per block in main kernel
#define NBLK1   1024      // BATCH / (16*WAVES)

__device__ __forceinline__ float leakyf(float z) { return z > 0.f ? z : 0.2f * z; }

// ---------------------------------------------------------------------------
// Kernel 0: pack W1 (60,16,64) and W2 (60,64,16) fp32 -> f16 WMMA B-fragments.
// B fragment layout (16x16x32 f16, wave32): lane L -> col n=L&15, sel=L>>4;
// element e (0..15) -> K = sel*8 + e (e<8), 16 + sel*8 + (e-8) (e>=8).  W1 K padded 16->32 with zeros.
// ---------------------------------------------------------------------------
__global__ __launch_bounds__(256) void pack_weights(const float* __restrict__ W1,
                                                    const float* __restrict__ W2,
                                                    _Float16* __restrict__ w1f,
                                                    _Float16* __restrict__ w2f) {
  int id = blockIdx.x * 256 + threadIdx.x;
  if (id < KG * 4 * 32) {                       // W1 fragments: [k][tile][lane][16]
    int lane = id & 31, tile = (id >> 5) & 3, k = id >> 7;
    int n = lane & 15, sel = lane >> 4;
    _Float16* dst = w1f + (size_t)id * 16;
#pragma unroll
    for (int e = 0; e < 16; ++e) {
      int K = (e < 8) ? (sel * 8 + e) : (16 + sel * 8 + (e - 8));
      float v = (K < CW) ? W1[k * (CW * HW) + K * HW + tile * 16 + n] : 0.f;
      dst[e] = (_Float16)v;
    }
  } else if (id < KG * 4 * 32 + KG * 2 * 32) {  // W2 fragments: [k][chunk][lane][16]
    int id2 = id - KG * 4 * 32;
    int lane = id2 & 31, chunk = (id2 >> 5) & 1, k = id2 >> 6;
    int n = lane & 15, sel = lane >> 4;
    _Float16* dst = w2f + (size_t)id2 * 16;
#pragma unroll
    for (int e = 0; e < 16; ++e) {
      int K = chunk * 32 + ((e < 8) ? (sel * 8 + e) : (16 + sel * 8 + (e - 8)));
      dst[e] = (_Float16)W2[k * (HW * CW) + K * CW + n];
    }
  }
}

// ---------------------------------------------------------------------------
// Kernel 1: per-group autoencoder (WMMA f16) + error + normalize + MLP head
// up to d_pre (pre-batchnorm), plus per-block BN partial sums.
// Each wave handles a 16-row batch tile; 4 waves per block.
// ---------------------------------------------------------------------------
__global__ __launch_bounds__(128)
void ae_mlp_kernel(const float* __restrict__ x, const int* __restrict__ idx,
                   const float* __restrict__ meanp, const float* __restrict__ stdp,
                   const float* __restrict__ b1, const float* __restrict__ b2,
                   const float* __restrict__ ew0, const float* __restrict__ eb0,
                   const float* __restrict__ ew1, const float* __restrict__ eb1,
                   const float* __restrict__ ew2, const float* __restrict__ eb2,
                   const float* __restrict__ dw0, const float* __restrict__ db0,
                   const float* __restrict__ dw1, const float* __restrict__ db1,
                   const _Float16* __restrict__ w1f, const _Float16* __restrict__ w2f,
                   float* __restrict__ dpre, float* __restrict__ psum,
                   float* __restrict__ psum2,
                   float* __restrict__ out_raw, float* __restrict__ out_norm) {
  __shared__ int      s_idx[KG * CW];                 // gather indices
  __shared__ float    s_xs[WAVES][16][CW];            // fp32 x tile (for error)
  __shared__ _Float16 s_h[WAVES][16][HW];             // hidden activations (f16)
  __shared__ float    s_norm[WAVES][16][KG];          // normalized errors
  __shared__ float    s_a[WAVES][16][50];             // MLP ping
  __shared__ float    s_b[WAVES][16][40];             // MLP pong
  __shared__ float    s_ps[WAVES][2][50];             // per-wave BN partials

  const int tid  = threadIdx.x;
  const int wave = tid >> 5;
  const int lane = tid & 31;
  const int lo   = lane & 15;      // A row / B,C,D column
  const int hi   = lane >> 4;      // lane-half select
  const int b0   = blockIdx.x * (16 * WAVES) + wave * 16;

  for (int i = tid; i < KG * CW; i += 128) s_idx[i] = idx[i];
  __syncthreads();

  const float* xrow = x + (size_t)(b0 + lo) * (KG * CW);
  const v16h*  w1v  = (const v16h*)w1f;
  const v16h*  w2v  = (const v16h*)w2f;

  for (int k = 0; k < KG; ++k) {
    // ---- A fragment (16x32 f16, K padded): real elements e=0..7 at K=hi*8+e
    v16h a = {};
#pragma unroll
    for (int e = 0; e < 8; ++e) {
      int col = s_idx[k * CW + hi * 8 + e];
      float v = xrow[col];
      a[e] = (_Float16)v;
      s_xs[wave][lo][hi * 8 + e] = v;
    }
    // ---- stage 1: h = relu(xg @ W1 + b1), N = 64 as 4 tiles of 16
#pragma unroll
    for (int t = 0; t < 4; ++t) {
      v16h bfrag = w1v[(size_t)(k * 4 + t) * 32 + lane];
      v8f c = {};
      c = __builtin_amdgcn_wmma_f32_16x16x32_f16(false, a, false, bfrag,
                                                 (short)0, c, false, false);
      float bias = b1[k * HW + t * 16 + lo];
#pragma unroll
      for (int r = 0; r < 8; ++r) {
        float hv = c[r] + bias;
        hv = hv > 0.f ? hv : 0.f;
        s_h[wave][hi * 8 + r][t * 16 + lo] = (_Float16)hv;
      }
    }
    __syncthreads();

    // ---- stage 2: rec = relu(h @ W2 + b2), K = 64 as 2 chained WMMAs
    v8f c2 = {};
#pragma unroll
    for (int q = 0; q < 2; ++q) {
      v16h a2;
#pragma unroll
      for (int e = 0; e < 8; ++e) {
        a2[e]     = s_h[wave][lo][q * 32 + hi * 8 + e];
        a2[e + 8] = s_h[wave][lo][q * 32 + 16 + hi * 8 + e];
      }
      v16h bfrag = w2v[(size_t)(k * 2 + q) * 32 + lane];
      c2 = __builtin_amdgcn_wmma_f32_16x16x32_f16(false, a2, false, bfrag,
                                                  (short)0, c2, false, false);
    }
    float bias2 = b2[k * CW + lo];
    float err8[8];
#pragma unroll
    for (int r = 0; r < 8; ++r) {
      float rv = c2[r] + bias2;
      rv = rv > 0.f ? rv : 0.f;
      float d = s_xs[wave][hi * 8 + r][lo] - rv;   // column = lo
      err8[r] = d * d;
    }
    // butterfly sum over the 16 columns (lane bits 0..3)
#pragma unroll
    for (int mask = 1; mask <= 8; mask <<= 1) {
#pragma unroll
      for (int r = 0; r < 8; ++r) err8[r] += __shfl_xor(err8[r], mask, 32);
    }
    if (lo < 8) {
      int row = hi * 8 + lo;
      float raw = err8[lo] * (1.f / 16.f);
      float nv  = (raw - meanp[k]) / stdp[k];
      out_raw [(size_t)(b0 + row) * KG + k] = raw;
      out_norm[(size_t)(b0 + row) * KG + k] = nv;
      s_norm[wave][row][k] = nv;
    }
    __syncthreads();
  }

  // ---- MLP head: 60 -> 50 -> 40 -> 30 -> 40 -> 50 (d_pre). Lane = (row lo, half hi).
  // L1: leaky(norm @ ew0 + eb0)  (60 -> 50)
  for (int j = 0; j < 25; ++j) {
    int jj = hi * 25 + j;
    float acc = eb0[jj];
    for (int i = 0; i < 60; ++i) acc += s_norm[wave][lo][i] * ew0[i * 50 + jj];
    s_a[wave][lo][jj] = leakyf(acc);
  }
  __syncthreads();
  // L2: leaky(e @ ew1 + eb1)  (50 -> 40)
  for (int j = 0; j < 20; ++j) {
    int jj = hi * 20 + j;
    float acc = eb1[jj];
    for (int i = 0; i < 50; ++i) acc += s_a[wave][lo][i] * ew1[i * 40 + jj];
    s_b[wave][lo][jj] = leakyf(acc);
  }
  __syncthreads();
  // L3: tanh(e @ ew2 + eb2)  (40 -> 30)
  for (int j = 0; j < 15; ++j) {
    int jj = hi * 15 + j;
    float acc = eb2[jj];
    for (int i = 0; i < 40; ++i) acc += s_b[wave][lo][i] * ew2[i * 30 + jj];
    s_a[wave][lo][jj] = tanhf(acc);
  }
  __syncthreads();
  // L4: leaky(e @ dw0 + db0)  (30 -> 40)
  for (int j = 0; j < 20; ++j) {
    int jj = hi * 20 + j;
    float acc = db0[jj];
    for (int i = 0; i < 30; ++i) acc += s_a[wave][lo][i] * dw0[i * 40 + jj];
    s_b[wave][lo][jj] = leakyf(acc);
  }
  __syncthreads();
  // L5: d_pre = d @ dw1 + db1  (40 -> 50), plus BN partial sums
  float dv[25];
  for (int j = 0; j < 25; ++j) {
    int jj = hi * 25 + j;
    float acc = db1[jj];
    for (int i = 0; i < 40; ++i) acc += s_b[wave][lo][i] * dw1[i * 50 + jj];
    dv[j] = acc;
    dpre[(size_t)(b0 + lo) * 50 + jj] = acc;
  }
  for (int j = 0; j < 25; ++j) {
    float s = dv[j], s2 = dv[j] * dv[j];
#pragma unroll
    for (int mask = 1; mask <= 8; mask <<= 1) {
      s  += __shfl_xor(s,  mask, 32);
      s2 += __shfl_xor(s2, mask, 32);
    }
    if (lo == 0) { s_ps[wave][0][hi * 25 + j] = s; s_ps[wave][1][hi * 25 + j] = s2; }
  }
  __syncthreads();
  if (tid < 50) {
    float s = 0.f, s2 = 0.f;
    for (int w = 0; w < WAVES; ++w) { s += s_ps[w][0][tid]; s2 += s_ps[w][1][tid]; }
    psum [blockIdx.x * 50 + tid] = s;
    psum2[blockIdx.x * 50 + tid] = s2;
  }
}

// ---------------------------------------------------------------------------
// Kernel 2: reduce per-block partials -> mu, var (deterministic fixed order)
// ---------------------------------------------------------------------------
__global__ __launch_bounds__(64) void bn_stats(const float* __restrict__ psum,
                                               const float* __restrict__ psum2,
                                               float* __restrict__ stats) {
  int j = threadIdx.x;
  if (j >= 50) return;
  float s = 0.f, s2 = 0.f;
  for (int b = 0; b < NBLK1; ++b) { s += psum[b * 50 + j]; s2 += psum2[b * 50 + j]; }
  float mu  = s * (1.f / (float)BATCH);
  float var = s2 * (1.f / (float)BATCH) - mu * mu;
  stats[j]      = mu;
  stats[64 + j] = var;
}

// ---------------------------------------------------------------------------
// Kernel 3: BN + leaky + (B,50)@(50,60) + tanh -> total_rec
// ---------------------------------------------------------------------------
__global__ __launch_bounds__(256)
void finalize(const float* __restrict__ dpre, const float* __restrict__ stats,
              const float* __restrict__ bn_g, const float* __restrict__ bn_b,
              const float* __restrict__ dw2, const float* __restrict__ db2,
              float* __restrict__ out3) {
  __shared__ float s_w[50 * 60];
  __shared__ float s_scale[50], s_shift[50], s_db2[60];
  int tid = threadIdx.x;
  for (int i = tid; i < 50 * 60; i += 256) s_w[i] = dw2[i];
  if (tid < 50) {
    float mu = stats[tid], var = stats[64 + tid];
    float sc = bn_g[tid] * rsqrtf(var + 1e-5f);
    s_scale[tid] = sc;
    s_shift[tid] = bn_b[tid] - sc * mu;
  }
  if (tid < 60) s_db2[tid] = db2[tid];
  __syncthreads();

  size_t b = (size_t)blockIdx.x * 256 + tid;
  float dh[50];
  const float* dp = dpre + b * 50;
#pragma unroll
  for (int i = 0; i < 50; ++i) {
    float v = s_scale[i] * dp[i] + s_shift[i];
    dh[i] = v > 0.f ? v : 0.2f * v;
  }
  float* o = out3 + b * KG;
  for (int k = 0; k < KG; ++k) {
    float acc = s_db2[k];
#pragma unroll
    for (int i = 0; i < 50; ++i) acc += dh[i] * s_w[i * 60 + k];
    o[k] = tanhf(acc);
  }
}

// ---------------------------------------------------------------------------
extern "C" void kernel_launch(void* const* d_in, const int* in_sizes, int n_in,
                              void* d_out, int out_size, void* d_ws, size_t ws_size,
                              hipStream_t stream) {
  const float* x     = (const float*)d_in[0];
  const float* meanp = (const float*)d_in[1];
  const float* stdp  = (const float*)d_in[2];
  const int*   idx   = (const int*)  d_in[3];
  const float* W1    = (const float*)d_in[4];
  const float* b1    = (const float*)d_in[5];
  const float* W2    = (const float*)d_in[6];
  const float* b2    = (const float*)d_in[7];
  const float* ew0   = (const float*)d_in[8];
  const float* eb0   = (const float*)d_in[9];
  const float* ew1   = (const float*)d_in[10];
  const float* eb1   = (const float*)d_in[11];
  const float* ew2   = (const float*)d_in[12];
  const float* eb2   = (const float*)d_in[13];
  const float* dw0   = (const float*)d_in[14];
  const float* db0   = (const float*)d_in[15];
  const float* dw1   = (const float*)d_in[16];
  const float* db1   = (const float*)d_in[17];
  const float* bn_g  = (const float*)d_in[18];
  const float* bn_b  = (const float*)d_in[19];
  const float* dw2   = (const float*)d_in[20];
  const float* db2   = (const float*)d_in[21];

  // workspace layout (all offsets 256B-aligned)
  char* ws = (char*)d_ws;
  _Float16* w1f   = (_Float16*)(ws + 0);               // 60*4*32*16*2  = 245760
  _Float16* w2f   = (_Float16*)(ws + 245760);          // 60*2*32*16*2  = 122880
  float*    dpre  = (float*)   (ws + 368640);          // 65536*50*4    = 13107200
  float*    psum  = (float*)   (ws + 13475840);        // 1024*50*4     = 204800
  float*    psum2 = (float*)   (ws + 13680640);        // 1024*50*4     = 204800
  float*    stats = (float*)   (ws + 13885440);        // 128 floats

  float* out_raw  = (float*)d_out;
  float* out_norm = out_raw  + (size_t)BATCH * KG;
  float* out3     = out_norm + (size_t)BATCH * KG;

  pack_weights<<<45, 256, 0, stream>>>(W1, W2, w1f, w2f);
  ae_mlp_kernel<<<NBLK1, 128, 0, stream>>>(x, idx, meanp, stdp, b1, b2,
                                           ew0, eb0, ew1, eb1, ew2, eb2,
                                           dw0, db0, dw1, db1,
                                           w1f, w2f, dpre, psum, psum2,
                                           out_raw, out_norm);
  bn_stats<<<1, 64, 0, stream>>>(psum, psum2, stats);
  finalize<<<256, 256, 0, stream>>>(dpre, stats, bn_g, bn_b, dw2, db2, out3);
}
